// BMEDLSTM_46359876993056
// MI455X (gfx1250) — matrix operand order repaired and checked by reading, not compiled, not run on hardware
//
#include <hip/hip_runtime.h>
#include <hip/hip_bf16.h>

typedef __attribute__((ext_vector_type(16))) _Float16 v16h;
typedef __attribute__((ext_vector_type(8)))  _Float16 v8h;
typedef __attribute__((ext_vector_type(8)))  float    v8f;

#define DT_STEP 0.01f
#define WAVES   4          // waves per block
#define BLOCKS  64         // 64 * 4 * 16 = 4096 batch rows
#define BATCH   4096

// LDS layout (halves unless noted)
#define W0_N 256
#define W0_K 96            // 16 (x|s) + 64 (h_a) + 16 pad
#define W1_N 256
#define W1_K 128           // 64 (h_a) + 64 (h_b)
#define WF_N 16            // 8 real rows + 8 zero pad
#define WF_K 64

#define SM_W0   0
#define SM_W1   (SM_W0 + W0_N*W0_K)
#define SM_WF   (SM_W1 + W1_N*W1_K)
#define SM_B0   (SM_WF + WF_N*WF_K)
#define SM_BIAS_HALVES (2*(256+256+16))
#define SM_STAGE (SM_B0 + SM_BIAS_HALVES)
#define STAGE_PER_WAVE (16*W0_K + 16*W1_K)
#define SMEM_HALVES (SM_STAGE + WAVES*STAGE_PER_WAVE)
#define SMEM_BYTES  (SMEM_HALVES*2)

// ---- fragment loaders matching CDNA5 WMMA VGPR layouts (16-bit, wave32) ----
// A 16x32: lane m=l&15, hi=l>>4; halves 0-7 -> K = kbase+8*hi..+7
//                                halves 8-15 -> K = kbase+16+8*hi..+7
__device__ __forceinline__ v16h load_afrag(const _Float16* base, int strideH,
                                           int kbase, int lane) {
  const int m  = lane & 15;
  const int hi = lane >> 4;
  const _Float16* p = base + m * strideH + kbase + 8 * hi;
  union { v16h v; v8h h[2]; } u;
  u.h[0] = *(const v8h*)(p);
  u.h[1] = *(const v8h*)(p + 16);
  return u.v;
}

// B 32x16: lane n=(l&15)+16*ntile in weight rows, halves 0-7 -> K = kbase+8*hi..+7, halves 8-15 -> K = kbase+16+8*hi..+7
// weights stored row-major [n][k] (row n = output/gate index), stride strideH halves
__device__ __forceinline__ v16h load_bfrag(const _Float16* w, int strideH,
                                           int ntile, int kbase, int lane) {
  const int n  = (lane & 15) + 16 * ntile;
  const int hi = lane >> 4;
  const _Float16* p = w + n * strideH + kbase + 8 * hi;
  union { v16h v; v8h h[2]; } u;
  u.h[0] = *(const v8h*)(p);
  u.h[1] = *(const v8h*)(p + 16);
  return u.v;
}

// CDNA5 hardware tanh (TRANS pipe). v_nop inside the asm covers the
// documented TRANS-result hazard (1 independent op before use), since the
// scheduler cannot see through inline asm.
__device__ __forceinline__ float tanh_hw(float x) {
  float y;
  asm("v_tanh_f32 %0, %1\n\tv_nop" : "=v"(y) : "v"(x));
  return y;
}
// sigmoid(x) = 0.5*tanh(0.5x) + 0.5  -> mul + v_tanh + fma
__device__ __forceinline__ float sig_hw(float x) {
  return __builtin_fmaf(0.5f, tanh_hw(0.5f * x), 0.5f);
}

__device__ __forceinline__ v8f splat8(float s) {
  v8f r;
#pragma unroll
  for (int i = 0; i < 8; ++i) r[i] = s;
  return r;
}

__device__ __forceinline__ v8f wmma_f16(v16h a, v16h b, v8f c) {
  return __builtin_amdgcn_wmma_f32_16x16x32_f16(false, a, false, b,
                                                (short)0, c, false, false);
}

__global__ void __launch_bounds__(WAVES * 32)
lstm2_rollout_kernel(const float* __restrict__ x,     const float* __restrict__ s0,
                     const float* __restrict__ Wih0,  const float* __restrict__ Whh0,
                     const float* __restrict__ bih0,  const float* __restrict__ bhh0,
                     const float* __restrict__ Wih1,  const float* __restrict__ Whh1,
                     const float* __restrict__ bih1,  const float* __restrict__ bhh1,
                     const float* __restrict__ fcW,   const float* __restrict__ fcb,
                     float* __restrict__ out_outputs, float* __restrict__ out_times,
                     float* __restrict__ out_res,     int T)
{
  extern __shared__ _Float16 smem[];
  _Float16* w0 = smem + SM_W0;
  _Float16* w1 = smem + SM_W1;
  _Float16* wf = smem + SM_WF;
  float*    b0 = (float*)(smem + SM_B0);
  float*    b1 = b0 + 256;
  float*    bf = b1 + 256;

  const int tid  = threadIdx.x;
  const int wave = tid >> 5;
  const int lane = tid & 31;
  const int n_l  = lane & 15;
  const int hi_l = lane >> 4;

  _Float16* A0 = smem + SM_STAGE + wave * STAGE_PER_WAVE; // 16 x 96  : [x|s|h_a|0]
  _Float16* A1 = A0 + 16 * W0_K;                          // 16 x 128 : [h_a|h_b]

  // ---- cooperative weight load (f32 -> f16 into LDS, fused layouts) ----
  for (int i = tid; i < W0_N * W0_K; i += blockDim.x) {
    int n = i / W0_K, k = i % W0_K;
    float v = (k < 16) ? Wih0[n * 16 + k]
            : (k < 80) ? Whh0[n * 64 + (k - 16)] : 0.0f;
    w0[i] = (_Float16)v;
  }
  for (int i = tid; i < W1_N * W1_K; i += blockDim.x) {
    int n = i >> 7, k = i & 127;
    float v = (k < 64) ? Wih1[n * 64 + k] : Whh1[n * 64 + (k - 64)];
    w1[i] = (_Float16)v;
  }
  for (int i = tid; i < WF_N * WF_K; i += blockDim.x) {
    int n = i >> 6, k = i & 63;
    wf[i] = (_Float16)((n < 8) ? fcW[n * 64 + k] : 0.0f);
  }
  for (int i = tid; i < 256; i += blockDim.x) {
    b0[i] = bih0[i] + bhh0[i];
    b1[i] = bih1[i] + bhh1[i];
  }
  if (tid < 16) bf[tid] = (tid < 8) ? fcb[tid] : 0.0f;

  // cal_times written by block 0
  if (blockIdx.x == 0)
    for (int i = tid; i < T; i += blockDim.x) out_times[i] = (float)i * DT_STEP;

  // ---- per-wave staging init (zeros: h_a, h_b, pad) ----
  for (int i = lane; i < 16 * W0_K; i += 32) A0[i] = (_Float16)0.0f;
  for (int i = lane; i < 16 * W1_K; i += 32) A1[i] = (_Float16)0.0f;

  const int rowbase = (blockIdx.x * WAVES + wave) * 16;

  // xin -> A0 cols 0..7 (constant over all steps)
  for (int i = lane; i < 16 * 8; i += 32) {
    int m = i >> 3, j = i & 7;
    A0[m * W0_K + j] = (_Float16)x[(size_t)(rowbase + m) * 8 + j];
  }

  // s state in registers: lane (n_l<8) holds s[rowbase+v+8*hi][n_l] for v=0..7
  float sreg[8];
#pragma unroll
  for (int v = 0; v < 8; ++v) {
    int m = v + 8 * hi_l;
    sreg[v] = (n_l < 8) ? s0[(size_t)(rowbase + m) * 8 + n_l] : 0.0f;
  }
  if (n_l < 8)
#pragma unroll
    for (int v = 0; v < 8; ++v) {
      int m = v + 8 * hi_l;
      A0[m * W0_K + 8 + n_l] = (_Float16)sreg[v];
    }

  v8f ca[4], cb[4];
#pragma unroll
  for (int t = 0; t < 4; ++t)
#pragma unroll
    for (int v = 0; v < 8; ++v) { ca[t][v] = 0.0f; cb[t][v] = 0.0f; }

  __syncthreads();  // weights + biases visible

  // bias values, per-lane (n = n_l + 16*t)
  float rb0[16], rb1[16];
#pragma unroll
  for (int t = 0; t < 16; ++t) { rb0[t] = b0[n_l + 16 * t]; rb1[t] = b1[n_l + 16 * t]; }
  const float rbf = bf[n_l];

  const size_t rowstride = (size_t)T * 8;

  for (int step = 0; step < T; ++step) {
    // ============ layer 0: gates0 = [x|s|h_a] x W0^T + b0 ============
    // A fragments loaded once; gate tiles processed in i/f/g/o groups of 4
    // so only 4 accumulators are live at a time.
    v16h a00 = load_afrag(A0, W0_K, 0,  lane);
    v16h a01 = load_afrag(A0, W0_K, 32, lane);
    v16h a02 = load_afrag(A0, W0_K, 64, lane);
#pragma unroll
    for (int g = 0; g < 4; ++g) {
      v8f ai = splat8(rb0[g]);
      v8f af = splat8(rb0[g + 4]);
      v8f ag = splat8(rb0[g + 8]);
      v8f ao = splat8(rb0[g + 12]);
      ai = wmma_f16(a00, load_bfrag(w0, W0_K, g,      0,  lane), ai);
      af = wmma_f16(a00, load_bfrag(w0, W0_K, g + 4,  0,  lane), af);
      ag = wmma_f16(a00, load_bfrag(w0, W0_K, g + 8,  0,  lane), ag);
      ao = wmma_f16(a00, load_bfrag(w0, W0_K, g + 12, 0,  lane), ao);
      ai = wmma_f16(a01, load_bfrag(w0, W0_K, g,      32, lane), ai);
      af = wmma_f16(a01, load_bfrag(w0, W0_K, g + 4,  32, lane), af);
      ag = wmma_f16(a01, load_bfrag(w0, W0_K, g + 8,  32, lane), ag);
      ao = wmma_f16(a01, load_bfrag(w0, W0_K, g + 12, 32, lane), ao);
      ai = wmma_f16(a02, load_bfrag(w0, W0_K, g,      64, lane), ai);
      af = wmma_f16(a02, load_bfrag(w0, W0_K, g + 4,  64, lane), af);
      ag = wmma_f16(a02, load_bfrag(w0, W0_K, g + 8,  64, lane), ag);
      ao = wmma_f16(a02, load_bfrag(w0, W0_K, g + 12, 64, lane), ao);

      const int cA0 = 16 + n_l + 16 * g;
      const int cA1 = n_l + 16 * g;
#pragma unroll
      for (int v = 0; v < 8; ++v) {
        float ig = sig_hw(ai[v]);
        float fg = sig_hw(af[v]);
        float gg = tanh_hw(ag[v]);
        float og = sig_hw(ao[v]);
        float c  = fg * ca[g][v] + ig * gg;
        ca[g][v] = c;
        _Float16 hh = (_Float16)(og * tanh_hw(c));
        int m = v + 8 * hi_l;
        A0[m * W0_K + cA0] = hh;   // h_a for next step's layer 0
        A1[m * W1_K + cA1] = hh;   // h_a for this step's layer 1
      }
    }

    // ============ layer 1: gates1 = [h_a|h_b] x W1^T + b1 ============
    v16h a10 = load_afrag(A1, W1_K, 0,  lane);  // new h_a
    v16h a11 = load_afrag(A1, W1_K, 32, lane);
    v16h a12 = load_afrag(A1, W1_K, 64, lane);  // old h_b
    v16h a13 = load_afrag(A1, W1_K, 96, lane);  // old h_b
#pragma unroll
    for (int g = 0; g < 4; ++g) {
      v8f ai = splat8(rb1[g]);
      v8f af = splat8(rb1[g + 4]);
      v8f ag = splat8(rb1[g + 8]);
      v8f ao = splat8(rb1[g + 12]);
      ai = wmma_f16(a10, load_bfrag(w1, W1_K, g,      0,  lane), ai);
      af = wmma_f16(a10, load_bfrag(w1, W1_K, g + 4,  0,  lane), af);
      ag = wmma_f16(a10, load_bfrag(w1, W1_K, g + 8,  0,  lane), ag);
      ao = wmma_f16(a10, load_bfrag(w1, W1_K, g + 12, 0,  lane), ao);
      ai = wmma_f16(a11, load_bfrag(w1, W1_K, g,      32, lane), ai);
      af = wmma_f16(a11, load_bfrag(w1, W1_K, g + 4,  32, lane), af);
      ag = wmma_f16(a11, load_bfrag(w1, W1_K, g + 8,  32, lane), ag);
      ao = wmma_f16(a11, load_bfrag(w1, W1_K, g + 12, 32, lane), ao);
      ai = wmma_f16(a12, load_bfrag(w1, W1_K, g,      64, lane), ai);
      af = wmma_f16(a12, load_bfrag(w1, W1_K, g + 4,  64, lane), af);
      ag = wmma_f16(a12, load_bfrag(w1, W1_K, g + 8,  64, lane), ag);
      ao = wmma_f16(a12, load_bfrag(w1, W1_K, g + 12, 64, lane), ao);
      ai = wmma_f16(a13, load_bfrag(w1, W1_K, g,      96, lane), ai);
      af = wmma_f16(a13, load_bfrag(w1, W1_K, g + 4,  96, lane), af);
      ag = wmma_f16(a13, load_bfrag(w1, W1_K, g + 8,  96, lane), ag);
      ao = wmma_f16(a13, load_bfrag(w1, W1_K, g + 12, 96, lane), ao);

      const int cA1 = 64 + n_l + 16 * g;
#pragma unroll
      for (int v = 0; v < 8; ++v) {
        float ig = sig_hw(ai[v]);
        float fg = sig_hw(af[v]);
        float gg = tanh_hw(ag[v]);
        float og = sig_hw(ao[v]);
        float c  = fg * cb[g][v] + ig * gg;
        cb[g][v] = c;
        int m = v + 8 * hi_l;
        A1[m * W1_K + cA1] = (_Float16)(og * tanh_hw(c));  // h_b
      }
    }

    // ============ fc: step_out = h_b x fcW^T + fcb ============
    v16h af0 = load_afrag(A1, W1_K, 64, lane);  // new h_b
    v16h af1 = load_afrag(A1, W1_K, 96, lane);
    v8f accf = splat8(rbf);
    accf = wmma_f16(af0, load_bfrag(wf, WF_K, 0, 0,  lane), accf);
    accf = wmma_f16(af1, load_bfrag(wf, WF_K, 0, 32, lane), accf);

    // s integration + global stores (only lanes with n_l < 8 hold valid columns)
    if (n_l < 8) {
#pragma unroll
      for (int v = 0; v < 8; ++v) {
        int m = v + 8 * hi_l;
        float so = accf[v];
        float sn = sreg[v] + DT_STEP * so;
        sreg[v] = sn;
        size_t idx = (size_t)(rowbase + m) * rowstride + (size_t)step * 8 + n_l;
        out_outputs[idx] = sn;
        out_res[idx]     = so;
        A0[m * W0_K + 8 + n_l] = (_Float16)sn;  // feed s into next step's A operand
      }
    }
  }
}

extern "C" void kernel_launch(void* const* d_in, const int* in_sizes, int n_in,
                              void* d_out, int out_size, void* d_ws, size_t ws_size,
                              hipStream_t stream) {
  const float* x    = (const float*)d_in[0];
  const float* s0   = (const float*)d_in[1];
  const float* Wih0 = (const float*)d_in[2];
  const float* Whh0 = (const float*)d_in[3];
  const float* bih0 = (const float*)d_in[4];
  const float* bhh0 = (const float*)d_in[5];
  const float* Wih1 = (const float*)d_in[6];
  const float* Whh1 = (const float*)d_in[7];
  const float* bih1 = (const float*)d_in[8];
  const float* bhh1 = (const float*)d_in[9];
  const float* fcW  = (const float*)d_in[10];
  const float* fcb  = (const float*)d_in[11];
  // num_steps lives on device; recover T from out_size = T*(2*B*8 + 1)
  const int T = out_size / (2 * BATCH * 8 + 1);

  float* out_outputs = (float*)d_out;                       // (B, T, 8)
  float* out_times   = out_outputs + (size_t)BATCH * T * 8; // (T,)
  float* out_res     = out_times + T;                       // (B, T, 8)

  static bool attr_set = false;  // idempotent attribute set (not state that alters work)
  if (!attr_set) {
    hipFuncSetAttribute((const void*)lstm2_rollout_kernel,
                        hipFuncAttributeMaxDynamicSharedMemorySize, SMEM_BYTES);
    attr_set = true;
  }

  lstm2_rollout_kernel<<<BLOCKS, WAVES * 32, SMEM_BYTES, stream>>>(
      x, s0, Wih0, Whh0, bih0, bhh0, Wih1, Whh1, bih1, bhh1, fcW, fcb,
      out_outputs, out_times, out_res, T);
}
